// SOC_Mamba_V2_22230750724315
// MI455X (gfx1250) — compile-verified, matrix operand order checked by reading, not run on hardware
//
#include <hip/hip_runtime.h>
#include <hip/hip_bf16.h>

typedef __attribute__((ext_vector_type(16))) _Float16 v16h;
typedef __attribute__((ext_vector_type(8)))  _Float16 v8h;
typedef __attribute__((ext_vector_type(8)))  float    v8f;

#define DEV __device__ __forceinline__

constexpr int BQ = 8, TQ = 2048, BT = BQ * TQ;     // batch, time, tokens
constexpr int DM = 256, DI = 512, DS = 16, DR = 16, DC = 4;
constexpr int NXP = 64;                             // x_proj N padded 48 -> 64
constexpr int NC = 16, CS = TQ / NC;                // scan chunks
constexpr int ST = 16;                              // B/C LDS staging steps
constexpr float EPSF = 1e-5f;

DEV float siluf(float x)     { return x / (1.f + __expf(-x)); }
DEV float softplusf(float x) { return (x > 20.f) ? x : log1pf(__expf(x)); }

// ---------------- input projection + LayerNorm (K=10, scalar fp32) ----------
__global__ void k_input_ln(const float* __restrict__ x, const float* __restrict__ w,
                           const float* __restrict__ b, const float* __restrict__ g,
                           const float* __restrict__ be, float* __restrict__ h) {
  __shared__ float red[DM];
  int m = blockIdx.x, tid = threadIdx.x;
  const float* xr = x + (size_t)m * 10;
  const float* wr = w + tid * 10;
  float acc = b[tid];
#pragma unroll
  for (int j = 0; j < 10; j++) acc += xr[j] * wr[j];
  red[tid] = acc; __syncthreads();
  for (int s = 128; s > 0; s >>= 1) { if (tid < s) red[tid] += red[tid + s]; __syncthreads(); }
  float mean = red[0] / DM; __syncthreads();
  float dv = acc - mean;
  red[tid] = dv * dv; __syncthreads();
  for (int s = 128; s > 0; s >>= 1) { if (tid < s) red[tid] += red[tid + s]; __syncthreads(); }
  float var = red[0] / DM;
  h[(size_t)m * DM + tid] = dv * rsqrtf(var + EPSF) * g[tid] + be[tid];
}

// ---------------- RMSNorm + cast to f16 -------------------------------------
__global__ void k_rmsnorm_cast(const float* __restrict__ h, const float* __restrict__ w,
                               _Float16* __restrict__ out) {
  __shared__ float red[DM];
  int m = blockIdx.x, tid = threadIdx.x;
  float v = h[(size_t)m * DM + tid];
  red[tid] = v * v; __syncthreads();
  for (int s = 128; s > 0; s >>= 1) { if (tid < s) red[tid] += red[tid + s]; __syncthreads(); }
  float scale = rsqrtf(red[0] / DM + EPSF);
  out[(size_t)m * DM + tid] = (_Float16)(v * scale * w[tid]);
}

// ---------------- weight conversions to f16 ---------------------------------
__global__ void k_cvt_f16(const float* __restrict__ s, _Float16* __restrict__ d, int n) {
  int i = blockIdx.x * 256 + threadIdx.x;
  if (i < n) d[i] = (_Float16)s[i];
}
__global__ void k_cvt_f16_pad(const float* __restrict__ s, _Float16* __restrict__ d,
                              int rows, int cols, int dstRows) {
  int i = blockIdx.x * 256 + threadIdx.x;
  if (i >= dstRows * cols) return;
  int r = i / cols;
  d[i] = (r < rows) ? (_Float16)s[i] : (_Float16)0.f;
}

// ---------------- WMMA GEMM: C[M,N](f32) = A[M,K](f16) * W[N,K](f16)^T ------
// Block = 4 waves stacked in M; each wave computes a 16(M) x 64(N) tile.
// The 64xK weight slab (shared by all 4 waves) is staged into LDS once via
// async global->LDS b128 loads (ASYNCcnt), then B fragments come from LDS
// (ds_load_b128) while A streams from global as two b128 loads per k-step.
template <bool ACC, int K>
__global__ void k_gemm_wmma(const _Float16* __restrict__ A, const _Float16* __restrict__ W,
                            float* __restrict__ C, int M, int N) {
  __shared__ _Float16 sW[64 * K];
  const int tid  = threadIdx.x;
  const int lane = tid & 31;
  const int wave = tid >> 5;
  const int r  = lane & 15;
  const int hb = lane >> 4;
  const int m0 = (blockIdx.x * 4 + wave) * 16;
  const int n0 = blockIdx.y * 64;

  // ---- async-stage weight slab W[n0 .. n0+63, 0..K) into LDS --------------
  {
    const unsigned long long gbase = (unsigned long long)(W + (size_t)n0 * K);
    const unsigned lbase = (unsigned)(uintptr_t)(&sW[0]);
    constexpr int CHUNKS = (64 * K * 2) / 16;     // 16B chunks in slab
#pragma unroll
    for (int j = 0; j < CHUNKS / 128; j++) {
      const int i = tid + j * 128;
      unsigned laddr = lbase + (unsigned)i * 16u;
      unsigned long long gaddr = gbase + (unsigned long long)i * 16ull;
      asm volatile("global_load_async_to_lds_b128 %0, %1, off"
                   :: "v"(laddr), "v"(gaddr) : "memory");
    }
    asm volatile("s_wait_asynccnt 0x0" ::: "memory");
  }
  __syncthreads();

  v8f acc[4];
#pragma unroll
  for (int nt = 0; nt < 4; nt++) {
    if (ACC) {
#pragma unroll
      for (int i = 0; i < 8; i++)
        acc[nt][i] = C[(size_t)(m0 + hb * 8 + i) * N + n0 + nt * 16 + r];
    } else {
      v8f z = {};
      acc[nt] = z;
    }
  }

  const _Float16* arow = A + (size_t)(m0 + r) * K;
  for (int k0 = 0; k0 < K; k0 += 32) {
    v8h a_lo = *(const v8h*)(arow + k0 + hb * 8);
    v8h a_hi = *(const v8h*)(arow + k0 + 16 + hb * 8);
    v16h af = __builtin_shufflevector(a_lo, a_hi, 0,1,2,3,4,5,6,7,8,9,10,11,12,13,14,15);
#pragma unroll
    for (int nt = 0; nt < 4; nt++) {
      const _Float16* wrow = &sW[(nt * 16 + r) * K];
      v8h b_lo = *(const v8h*)(wrow + k0 + hb * 8);
      v8h b_hi = *(const v8h*)(wrow + k0 + 16 + hb * 8);
      v16h bf = __builtin_shufflevector(b_lo, b_hi, 0,1,2,3,4,5,6,7,8,9,10,11,12,13,14,15);
      acc[nt] = __builtin_amdgcn_wmma_f32_16x16x32_f16(
          false, af, false, bf, (short)0, acc[nt], false, false);
    }
  }

#pragma unroll
  for (int nt = 0; nt < 4; nt++)
#pragma unroll
    for (int i = 0; i < 8; i++)
      C[(size_t)(m0 + hb * 8 + i) * N + n0 + nt * 16 + r] = acc[nt][i];
}

// ---------------- causal depthwise conv (k=4) + SiLU ------------------------
__global__ void k_conv_silu(const float* __restrict__ xz, const float* __restrict__ cw,
                            const float* __restrict__ cb, float* __restrict__ xi,
                            _Float16* __restrict__ xih) {
  int idx = blockIdx.x * 256 + threadIdx.x;
  if (idx >= BT * DI) return;
  int d = idx & (DI - 1);
  int m = idx / DI;
  int t = m & (TQ - 1);
  const float* wd = cw + d * DC;
  float s = cb[d];
#pragma unroll
  for (int j = 0; j < DC; j++) {
    int tt = t - (DC - 1) + j;
    if (tt >= 0) s += wd[j] * xz[(size_t)(m - (DC - 1) + j) * (2 * DI) + d];
  }
  float v = siluf(s);
  xi[idx] = v;
  xih[idx] = (_Float16)v;
}

// ---------------- delta = softplus(dbc[:,:16] @ dtw^T + dtb) ----------------
__global__ void k_dt(const float* __restrict__ dbc, const float* __restrict__ dtw,
                     const float* __restrict__ dtb, float* __restrict__ delta) {
  int idx = blockIdx.x * 256 + threadIdx.x;
  if (idx >= BT * DI) return;
  int d = idx & (DI - 1);
  int m = idx / DI;
  const float* row = dbc + (size_t)m * NXP;
  const float* wd = dtw + d * DR;
  float a = dtb[d];
#pragma unroll
  for (int j = 0; j < DR; j++) a += row[j] * wd[j];
  delta[idx] = softplusf(a);
}

// ---------------- selective scan: 3-pass chunked parallel scan --------------
// Pass 1: per (b,d,chunk) compute chunk-local scan-from-zero S[16] and sum(delta).
__global__ void k_scan1(const float* __restrict__ delta, const float* __restrict__ xi,
                        const float* __restrict__ dbc, const float* __restrict__ A_log,
                        float* __restrict__ Sbuf, float* __restrict__ dsum) {
  __shared__ float sB[ST][DS];
  int tid = threadIdx.x;
  int bi = blockIdx.x;                 // ((b*NC)+c)*2 + db
  int db = bi & 1, c = (bi >> 1) & (NC - 1), b = bi >> 5;
  int d = db * 256 + tid;
  float Av[DS], h[DS];
#pragma unroll
  for (int n = 0; n < DS; n++) { Av[n] = -__expf(A_log[d * DS + n]); h[n] = 0.f; }
  float dsm = 0.f;
  size_t mb = (size_t)b * TQ + (size_t)c * CS;
  for (int t0 = 0; t0 < CS; t0 += ST) {
    __syncthreads();
    { int tt = tid >> 4, n = tid & 15;
      sB[tt][n] = dbc[(mb + t0 + tt) * NXP + DR + n]; }
    __syncthreads();
    for (int tt = 0; tt < ST; tt++) {
      size_t m = mb + t0 + tt;
      float dl = delta[m * DI + d];
      float xv = xi[m * DI + d];
      float dx = dl * xv;
      dsm += dl;
#pragma unroll
      for (int n = 0; n < DS; n++)
        h[n] = __expf(dl * Av[n]) * h[n] + dx * sB[tt][n];
    }
  }
  size_t base = ((size_t)(b * NC + c) * DS) * DI + d;
#pragma unroll
  for (int n = 0; n < DS; n++) Sbuf[base + (size_t)n * DI] = h[n];
  dsum[(size_t)(b * NC + c) * DI + d] = dsm;
}

// Pass 2: per (b,d) combine 16 chunk summaries -> start state for each chunk.
__global__ void k_scan2(const float* __restrict__ Sbuf, const float* __restrict__ dsum,
                        const float* __restrict__ A_log, float* __restrict__ hstart) {
  int idx = blockIdx.x * 256 + threadIdx.x;   // 0 .. BQ*DI-1
  int d = idx & (DI - 1), b = idx / DI;
  float Av[DS], h[DS];
#pragma unroll
  for (int n = 0; n < DS; n++) { Av[n] = -__expf(A_log[d * DS + n]); h[n] = 0.f; }
  for (int c = 0; c < NC; c++) {
    size_t base = ((size_t)(b * NC + c) * DS) * DI + d;
    float ds = dsum[(size_t)(b * NC + c) * DI + d];
#pragma unroll
    for (int n = 0; n < DS; n++) {
      hstart[base + (size_t)n * DI] = h[n];
      h[n] = __expf(Av[n] * ds) * h[n] + Sbuf[base + (size_t)n * DI];
    }
  }
}

// Pass 3: replay each chunk from its start state, emit y = (C.h + D*x)*silu(z) as f16.
__global__ void k_scan3(const float* __restrict__ delta, const float* __restrict__ xi,
                        const float* __restrict__ dbc, const float* __restrict__ A_log,
                        const float* __restrict__ Dp, const float* __restrict__ xz,
                        const float* __restrict__ hstart, _Float16* __restrict__ yh) {
  __shared__ float sB[ST][DS], sC[ST][DS];
  int tid = threadIdx.x;
  int bi = blockIdx.x;
  int db = bi & 1, c = (bi >> 1) & (NC - 1), b = bi >> 5;
  int d = db * 256 + tid;
  float Av[DS], h[DS];
  size_t base = ((size_t)(b * NC + c) * DS) * DI + d;
#pragma unroll
  for (int n = 0; n < DS; n++) {
    Av[n] = -__expf(A_log[d * DS + n]);
    h[n] = hstart[base + (size_t)n * DI];
  }
  float Dv = Dp[d];
  size_t mb = (size_t)b * TQ + (size_t)c * CS;
  for (int t0 = 0; t0 < CS; t0 += ST) {
    __syncthreads();
    { int tt = tid >> 4, n = tid & 15;
      sB[tt][n] = dbc[(mb + t0 + tt) * NXP + DR + n];
      sC[tt][n] = dbc[(mb + t0 + tt) * NXP + DR + DS + n]; }
    __syncthreads();
    for (int tt = 0; tt < ST; tt++) {
      size_t m = mb + t0 + tt;
      float dl = delta[m * DI + d];
      float xv = xi[m * DI + d];
      float dx = dl * xv;
      float y = 0.f;
#pragma unroll
      for (int n = 0; n < DS; n++) {
        h[n] = __expf(dl * Av[n]) * h[n] + dx * sB[tt][n];
        y += h[n] * sC[tt][n];
      }
      y += Dv * xv;
      float z = xz[m * (2 * DI) + DI + d];
      y *= siluf(z);
      yh[m * DI + d] = (_Float16)y;
    }
  }
}

// ---------------- final RMSNorm + head + past_soc add -----------------------
__global__ void k_head(const float* __restrict__ h, const float* __restrict__ frms,
                       const float* __restrict__ hw, const float* __restrict__ hb,
                       const float* __restrict__ x, float* __restrict__ out) {
  __shared__ float red[DM];
  int m = blockIdx.x, tid = threadIdx.x;
  float v = h[(size_t)m * DM + tid];
  red[tid] = v * v; __syncthreads();
  for (int s = 128; s > 0; s >>= 1) { if (tid < s) red[tid] += red[tid + s]; __syncthreads(); }
  float scale = rsqrtf(red[0] / DM + EPSF);
  __syncthreads();
  red[tid] = v * scale * frms[tid] * hw[tid]; __syncthreads();
  for (int s = 128; s > 0; s >>= 1) { if (tid < s) red[tid] += red[tid + s]; __syncthreads(); }
  if (tid == 0) {
    float dlt = red[0] + hb[0];
    out[BT + m] = dlt;                       // delta output
    out[m] = x[(size_t)m * 10 + 9] + dlt;    // y = past_soc + delta
  }
}

// ---------------------------------------------------------------------------
extern "C" void kernel_launch(void* const* d_in, const int* in_sizes, int n_in,
                              void* d_out, int out_size, void* d_ws, size_t ws_size,
                              hipStream_t stream) {
  (void)in_sizes; (void)n_in; (void)out_size; (void)ws_size;
  const float* x    = (const float*)d_in[0];
  const float* ipw  = (const float*)d_in[1];
  const float* ipb  = (const float*)d_in[2];
  const float* lng  = (const float*)d_in[3];
  const float* lnb  = (const float*)d_in[4];
  const float* inw  = (const float*)d_in[5];
  const float* cw   = (const float*)d_in[6];
  const float* cb   = (const float*)d_in[7];
  const float* xpw  = (const float*)d_in[8];
  const float* dtw  = (const float*)d_in[9];
  const float* dtb  = (const float*)d_in[10];
  const float* alog = (const float*)d_in[11];
  const float* Dp   = (const float*)d_in[12];
  const float* opw  = (const float*)d_in[13];
  const float* rmsw = (const float*)d_in[14];
  const float* frms = (const float*)d_in[15];
  const float* hw   = (const float*)d_in[16];
  const float* hb   = (const float*)d_in[17];
  float* out = (float*)d_out;

  char* p = (char*)d_ws;
  auto alloc = [&](size_t bytes) -> char* {
    char* q = p; p += (bytes + 255) & ~(size_t)255; return q;
  };
  float*    h    = (float*)alloc((size_t)BT * DM * 4);
  _Float16* t0h  = (_Float16*)alloc((size_t)BT * DM * 2);
  float*    xz   = (float*)alloc((size_t)BT * 2 * DI * 4);
  float*    xi   = (float*)alloc((size_t)BT * DI * 4);
  _Float16* xih  = (_Float16*)alloc((size_t)BT * DI * 2);
  float*    dbc  = (float*)alloc((size_t)BT * NXP * 4);
  float*    dlt  = (float*)alloc((size_t)BT * DI * 4);
  _Float16* yh   = (_Float16*)alloc((size_t)BT * DI * 2);
  float*    Sbuf = (float*)alloc((size_t)BQ * NC * DS * DI * 4);
  float*    dsum = (float*)alloc((size_t)BQ * NC * DI * 4);
  float*    hst  = (float*)alloc((size_t)BQ * NC * DS * DI * 4);
  _Float16* winh = (_Float16*)alloc((size_t)2 * DI * DM * 2);
  _Float16* wxh  = (_Float16*)alloc((size_t)NXP * DI * 2);
  _Float16* woh  = (_Float16*)alloc((size_t)DM * DI * 2);

  k_input_ln<<<BT, DM, 0, stream>>>(x, ipw, ipb, lng, lnb, h);

  for (int l = 0; l < 2; l++) {
    const float* inw_l = inw + (size_t)l * 2 * DI * DM;
    const float* xpw_l = xpw + (size_t)l * 48 * DI;
    const float* opw_l = opw + (size_t)l * DM * DI;

    k_cvt_f16<<<(2 * DI * DM + 255) / 256, 256, 0, stream>>>(inw_l, winh, 2 * DI * DM);
    k_cvt_f16_pad<<<(NXP * DI + 255) / 256, 256, 0, stream>>>(xpw_l, wxh, 48, DI, NXP);
    k_cvt_f16<<<(DM * DI + 255) / 256, 256, 0, stream>>>(opw_l, woh, DM * DI);

    k_rmsnorm_cast<<<BT, DM, 0, stream>>>(h, rmsw + l * DM, t0h);

    dim3 g1(BT / 64, (2 * DI) / 64);
    k_gemm_wmma<false, DM><<<g1, 128, 0, stream>>>(t0h, winh, xz, BT, 2 * DI);

    k_conv_silu<<<(BT * DI + 255) / 256, 256, 0, stream>>>(xz, cw + l * DI * DC, cb + l * DI, xi, xih);

    dim3 g2(BT / 64, NXP / 64);
    k_gemm_wmma<false, DI><<<g2, 128, 0, stream>>>(xih, wxh, dbc, BT, NXP);

    k_dt<<<(BT * DI + 255) / 256, 256, 0, stream>>>(dbc, dtw + l * DI * DR, dtb + l * DI, dlt);

    const float* al = alog + (size_t)l * DI * DS;
    k_scan1<<<BQ * NC * 2, 256, 0, stream>>>(dlt, xi, dbc, al, Sbuf, dsum);
    k_scan2<<<(BQ * DI) / 256, 256, 0, stream>>>(Sbuf, dsum, al, hst);
    k_scan3<<<BQ * NC * 2, 256, 0, stream>>>(dlt, xi, dbc, al, Dp + l * DI, xz, hst, yh);

    dim3 g3(BT / 64, DM / 64);
    k_gemm_wmma<true, DI><<<g3, 128, 0, stream>>>(yh, woh, h, BT, DM);  // fused residual add
  }

  k_head<<<BT, DM, 0, stream>>>(h, frms, hw, hb, x, out);
}
